// S4DLM_38817914421765
// MI455X (gfx1250) — compile-verified, matrix operand order checked by reading, not run on hardware
//
#include <hip/hip_runtime.h>
#include <hip/hip_bf16.h>
#include <math.h>

typedef __bf16 bf16;
typedef __attribute__((ext_vector_type(16))) bf16  v16bf;
typedef __attribute__((ext_vector_type(8)))  bf16  v8bf;
typedef __attribute__((ext_vector_type(8)))  float v8f;

#define VOCAB   32000
#define HDIM    768
#define NLAYER  6
#define NSTATE  64
#define NBATCH  2
#define SEQ     2048
#define ROWS    (NBATCH*SEQ)   // 4096
#define H4      (4*HDIM)       // 3072
#define LNEPS   1e-5f

// ---------------------------------------------------------------------------
// Elementwise helpers
// ---------------------------------------------------------------------------
__global__ __launch_bounds__(256) void f2bf_k(const float* __restrict__ in,
                                              bf16* __restrict__ out, int n) {
  int i = blockIdx.x * 256 + threadIdx.x;
  if (i < n) out[i] = (bf16)in[i];
}

__global__ __launch_bounds__(256) void addk(float* __restrict__ h,
                                            const float* __restrict__ x, int n) {
  int i = blockIdx.x * 256 + threadIdx.x;
  if (i < n) h[i] += x[i];
}

// ---------------------------------------------------------------------------
// LDS-tiled transpose+convert: W [K][N] f32 -> WT [N][K] bf16.
// 32x32 tiles; both global read and global write fully coalesced.
// ---------------------------------------------------------------------------
__global__ __launch_bounds__(256) void wtrans_tiled(const float* __restrict__ W,
                                                    bf16* __restrict__ WT,
                                                    int K, int N) {
  __shared__ float tile[32][33];          // +1 pad kills LDS bank conflicts
  const int kb = blockIdx.y * 32;
  const int nb = blockIdx.x * 32;
  const int tx = threadIdx.x & 31;
  const int ty = threadIdx.x >> 5;        // 8 rows per pass
#pragma unroll
  for (int i = 0; i < 32; i += 8)
    tile[ty + i][tx] = W[(size_t)(kb + ty + i) * N + (nb + tx)];
  __syncthreads();
#pragma unroll
  for (int i = 0; i < 32; i += 8)
    WT[(size_t)(nb + ty + i) * K + (kb + tx)] = (bf16)tile[tx][ty + i];
}

// ---------------------------------------------------------------------------
// Embedding gather: h[row,:] = embed[tokens[row],:]
// ---------------------------------------------------------------------------
__global__ __launch_bounds__(256) void gather_k(const int* __restrict__ tokens,
                                                const float* __restrict__ embed,
                                                float* __restrict__ h) {
  int row = blockIdx.x;
  int tok = tokens[row];
  const float* src = embed + (size_t)tok * HDIM;
  float* dst = h + (size_t)row * HDIM;
  for (int i = threadIdx.x; i < HDIM; i += 256) dst[i] = src[i];
}

// ---------------------------------------------------------------------------
// LayerNorm: one block per row; outf!=null -> f32 out, else bf16 out
// ---------------------------------------------------------------------------
__global__ __launch_bounds__(256) void layernorm_k(const float* __restrict__ X,
                                                   const float* __restrict__ w,
                                                   const float* __restrict__ b,
                                                   float* __restrict__ outf,
                                                   bf16* __restrict__ outb) {
  __shared__ float ssum[256];
  __shared__ float ssq[256];
  const int row = blockIdx.x;
  const int tid = threadIdx.x;
  const float* px = X + (size_t)row * HDIM;
  float s = 0.f, q = 0.f;
  for (int i = tid; i < HDIM; i += 256) {
    float v = px[i];
    s += v; q += v * v;
  }
  ssum[tid] = s; ssq[tid] = q;
  __syncthreads();
  for (int st = 128; st > 0; st >>= 1) {
    if (tid < st) { ssum[tid] += ssum[tid + st]; ssq[tid] += ssq[tid + st]; }
    __syncthreads();
  }
  const float mean = ssum[0] * (1.f / HDIM);
  const float var  = ssq[0] * (1.f / HDIM) - mean * mean;
  const float rinv = rsqrtf(var + LNEPS);
  for (int i = tid; i < HDIM; i += 256) {
    float v = (px[i] - mean) * rinv * w[i] + b[i];
    if (outf) outf[(size_t)row * HDIM + i] = v;
    else      outb[(size_t)row * HDIM + i] = (bf16)v;
  }
}

// ---------------------------------------------------------------------------
// S4D scan: one wave32 per (batch,channel); 2 states per lane; in-place on X.
// Unrolled x4 so the 4 broadcast loads and the 4 shuffle-reduction trees are
// independent chains (load/DS latency amortized); the true serial dependency
// is only the 2-FMA diagonal state update.
// ---------------------------------------------------------------------------
__device__ inline float wave_sum32(float y) {
  y += __shfl_xor(y, 16, 32);
  y += __shfl_xor(y, 8, 32);
  y += __shfl_xor(y, 4, 32);
  y += __shfl_xor(y, 2, 32);
  y += __shfl_xor(y, 1, 32);
  return y;
}

__global__ __launch_bounds__(256) void s4d_scan_k(float* __restrict__ X,
                                                  const float* __restrict__ A_log,
                                                  const float* __restrict__ Bm,
                                                  const float* __restrict__ Cm,
                                                  const float* __restrict__ log_dt) {
  const int gt   = blockIdx.x * 256 + threadIdx.x;
  const int wv   = gt >> 5;
  const int lane = gt & 31;
  if (wv >= NBATCH * HDIM) return;
  const int bb = wv / HDIM;
  const int hc = wv % HDIM;

  const float dt = expf(log_dt[hc]);
  const int n0 = lane, n1 = lane + 32;
  const float Ab0 = expf(-dt * expf(A_log[n0]));
  const float Ab1 = expf(-dt * expf(A_log[n1]));
  const float Bb0 = dt * Bm[(size_t)hc * NSTATE + n0];
  const float Bb1 = dt * Bm[(size_t)hc * NSTATE + n1];
  const float C0  = Cm[(size_t)hc * NSTATE + n0];
  const float C1  = Cm[(size_t)hc * NSTATE + n1];

  float s0 = 0.f, s1 = 0.f;
  float* px = X + (size_t)bb * SEQ * HDIM + hc;
  for (int t = 0; t < SEQ; t += 4) {
    // 4 independent broadcast loads issue back-to-back
    const float x0 = px[(size_t)(t + 0) * HDIM];
    const float x1 = px[(size_t)(t + 1) * HDIM];
    const float x2 = px[(size_t)(t + 2) * HDIM];
    const float x3 = px[(size_t)(t + 3) * HDIM];
    s0 = fmaf(Ab0, s0, Bb0 * x0); s1 = fmaf(Ab1, s1, Bb1 * x0);
    float y0 = C0 * s0 + C1 * s1;
    s0 = fmaf(Ab0, s0, Bb0 * x1); s1 = fmaf(Ab1, s1, Bb1 * x1);
    float y1 = C0 * s0 + C1 * s1;
    s0 = fmaf(Ab0, s0, Bb0 * x2); s1 = fmaf(Ab1, s1, Bb1 * x2);
    float y2 = C0 * s0 + C1 * s1;
    s0 = fmaf(Ab0, s0, Bb0 * x3); s1 = fmaf(Ab1, s1, Bb1 * x3);
    float y3 = C0 * s0 + C1 * s1;
    // 4 independent reduction trees -> shuffle latency overlaps
    y0 = wave_sum32(y0);
    y1 = wave_sum32(y1);
    y2 = wave_sum32(y2);
    y3 = wave_sum32(y3);
    if (lane == 0) {
      px[(size_t)(t + 0) * HDIM] = y0;
      px[(size_t)(t + 1) * HDIM] = y1;
      px[(size_t)(t + 2) * HDIM] = y2;
      px[(size_t)(t + 3) * HDIM] = y3;
    }
  }
}

// ---------------------------------------------------------------------------
// WMMA bf16 GEMM (TN): C[M,N] = A[M,K] * B[N,K]^T, both row-major K-contiguous
// Block: 256 thr = 8 waves (4x2); wave tile 32x64 (2x4 of 16x16 frags);
// block tile 128x128. 8 WMMAs per K-step from 6 fragment loads.
// epi: 0 -> outf = acc(+bias), nontemporal; 1 -> outb = bf16(gelu(acc+bias));
//      2 -> outf += acc(+bias)  (fused residual)
// ---------------------------------------------------------------------------
__device__ inline v16bf ldfrag(const bf16* p) {
  // elems 0..7 = K[kh .. kh+7], elems 8..15 = K[kh+16 .. kh+23]  (ISA striping)
  v8bf lo = *(const v8bf*)(p);
  v8bf hi = *(const v8bf*)(p + 16);
  v16bf r;
#pragma unroll
  for (int i = 0; i < 8; ++i) { r[i] = lo[i]; r[i + 8] = hi[i]; }
  return r;
}

__device__ inline v8f wmma_bf16(v16bf a, v16bf b, v8f c) {
  return __builtin_amdgcn_wmma_f32_16x16x32_bf16(false, a, false, b,
                                                 (short)0, c, false, false);
}

__global__ __launch_bounds__(256) void gemm_tn_wmma(const bf16* __restrict__ A,
                                                    const bf16* __restrict__ B,
                                                    const float* __restrict__ bias,
                                                    float* __restrict__ outf,
                                                    bf16* __restrict__ outb,
                                                    int M, int N, int K, int epi) {
  const int lane = threadIdx.x & 31;
  const int wv   = threadIdx.x >> 5;
  const int m0 = blockIdx.y * 128 + (wv >> 1) * 32;   // 4 wave-rows
  const int n0 = blockIdx.x * 128 + (wv & 1) * 64;    // 2 wave-cols

  const int kh = (lane >> 4) ? 8 : 0;                 // K-half select (ISA striping)
  const bf16* pA0 = A + (size_t)(m0 + (lane & 15)) * K + kh;
  const bf16* pA1 = pA0 + (size_t)16 * K;
  const bf16* pB0 = B + (size_t)(n0 + (lane & 15)) * K + kh;
  const bf16* pB1 = pB0 + (size_t)16 * K;
  const bf16* pB2 = pB0 + (size_t)32 * K;
  const bf16* pB3 = pB0 + (size_t)48 * K;

  v8f acc[2][4] = {};

  for (int kk = 0; kk < K; kk += 32) {
    __builtin_prefetch(pA0 + kk + 256, 0, 1);
    __builtin_prefetch(pA1 + kk + 256, 0, 1);
    __builtin_prefetch(pB0 + kk + 256, 0, 1);
    __builtin_prefetch(pB1 + kk + 256, 0, 1);
    __builtin_prefetch(pB2 + kk + 256, 0, 1);
    __builtin_prefetch(pB3 + kk + 256, 0, 1);
    v16bf a0 = ldfrag(pA0 + kk);
    v16bf a1 = ldfrag(pA1 + kk);
    v16bf b0 = ldfrag(pB0 + kk);
    v16bf b1 = ldfrag(pB1 + kk);
    v16bf b2 = ldfrag(pB2 + kk);
    v16bf b3 = ldfrag(pB3 + kk);
    acc[0][0] = wmma_bf16(a0, b0, acc[0][0]);
    acc[0][1] = wmma_bf16(a0, b1, acc[0][1]);
    acc[0][2] = wmma_bf16(a0, b2, acc[0][2]);
    acc[0][3] = wmma_bf16(a0, b3, acc[0][3]);
    acc[1][0] = wmma_bf16(a1, b0, acc[1][0]);
    acc[1][1] = wmma_bf16(a1, b1, acc[1][1]);
    acc[1][2] = wmma_bf16(a1, b2, acc[1][2]);
    acc[1][3] = wmma_bf16(a1, b3, acc[1][3]);
  }

  const int rbase = (lane >> 4) ? 8 : 0;  // C/D layout: VGPR v -> row v (+8 hi half)
  const int col0  = lane & 15;
#pragma unroll
  for (int mi = 0; mi < 2; ++mi) {
#pragma unroll
    for (int ni = 0; ni < 4; ++ni) {
      const int cc = n0 + ni * 16 + col0;
      const float bv = bias ? bias[cc] : 0.f;
#pragma unroll
      for (int v = 0; v < 8; ++v) {
        const int rr = m0 + mi * 16 + rbase + v;
        const size_t idx = (size_t)rr * N + cc;
        const float val = acc[mi][ni][v] + bv;
        if (epi == 0) {
          __builtin_nontemporal_store(val, &outf[idx]);   // logits: streamed out
        } else if (epi == 1) {
          const float g = 0.5f * val * (1.f + erff(val * 0.70710678f));
          outb[idx] = (bf16)g;
        } else {
          outf[idx] += val;
        }
      }
    }
  }
}

// ---------------------------------------------------------------------------
// Host launcher
// ---------------------------------------------------------------------------
extern "C" void kernel_launch(void* const* d_in, const int* in_sizes, int n_in,
                              void* d_out, int out_size, void* d_ws, size_t ws_size,
                              hipStream_t stream) {
  (void)in_sizes; (void)n_in; (void)out_size; (void)ws_size;
  const int*   tokens = (const int*)  d_in[0];
  const float* embed  = (const float*)d_in[1];
  const float* ln1_w  = (const float*)d_in[2];
  const float* ln1_b  = (const float*)d_in[3];
  const float* A_log  = (const float*)d_in[4];
  const float* Bm     = (const float*)d_in[5];
  const float* Cm     = (const float*)d_in[6];
  const float* log_dt = (const float*)d_in[7];
  const float* ln2_w  = (const float*)d_in[8];
  const float* ln2_b  = (const float*)d_in[9];
  const float* W1     = (const float*)d_in[10];
  const float* b1     = (const float*)d_in[11];
  const float* W2     = (const float*)d_in[12];
  const float* b2     = (const float*)d_in[13];
  const float* lnf_w  = (const float*)d_in[14];
  const float* lnf_b  = (const float*)d_in[15];
  float* out = (float*)d_out;

  // Workspace carve-up (all sizes 256B-aligned by construction)
  char* ws = (char*)d_ws;
  float* h   = (float*)ws;  ws += (size_t)ROWS * HDIM * 4;          // 12.6 MB
  float* x   = (float*)ws;  ws += (size_t)ROWS * HDIM * 4;          // 12.6 MB
  bf16* xbf  = (bf16*)ws;   ws += (size_t)ROWS * HDIM * 2;          //  6.3 MB
  bf16* z1bf = (bf16*)ws;   ws += (size_t)ROWS * H4 * 2;            // 25.2 MB
  bf16* W1T  = (bf16*)ws;   ws += (size_t)NLAYER * H4 * HDIM * 2;   // 28.3 MB
  bf16* W2T  = (bf16*)ws;   ws += (size_t)NLAYER * HDIM * H4 * 2;   // 28.3 MB
  bf16* embf = (bf16*)ws;   ws += (size_t)VOCAB * HDIM * 2;         // 49.2 MB

  // Pre-convert / pre-transpose weights (deterministic, every call)
  {
    int n = VOCAB * HDIM;
    f2bf_k<<<(n + 255) / 256, 256, 0, stream>>>(embed, embf, n);
  }
  for (int l = 0; l < NLAYER; ++l) {
    // W1: [768 x 3072] -> W1T [3072 x 768]
    wtrans_tiled<<<dim3(H4 / 32, HDIM / 32), 256, 0, stream>>>(
        W1 + (size_t)l * HDIM * H4, W1T + (size_t)l * H4 * HDIM, HDIM, H4);
    // W2: [3072 x 768] -> W2T [768 x 3072]
    wtrans_tiled<<<dim3(HDIM / 32, H4 / 32), 256, 0, stream>>>(
        W2 + (size_t)l * H4 * HDIM, W2T + (size_t)l * HDIM * H4, H4, HDIM);
  }

  // Embedding gather
  gather_k<<<ROWS, 256, 0, stream>>>(tokens, embed, h);

  const dim3 blk(256);
  for (int l = 0; l < NLAYER; ++l) {
    // LN1 -> x (f32)
    layernorm_k<<<ROWS, blk, 0, stream>>>(h, ln1_w + l * HDIM, ln1_b + l * HDIM,
                                          x, nullptr);
    // S4D scan in-place on x (1536 waves = 192 blocks exactly)
    s4d_scan_k<<<(NBATCH * HDIM * 32) / 256, blk, 0, stream>>>(
        x, A_log + l * NSTATE, Bm + (size_t)l * HDIM * NSTATE,
        Cm + (size_t)l * HDIM * NSTATE, log_dt + l * HDIM);
    // h += x
    addk<<<(ROWS * HDIM + 255) / 256, blk, 0, stream>>>(h, x, ROWS * HDIM);
    // LN2 -> xbf (bf16)
    layernorm_k<<<ROWS, blk, 0, stream>>>(h, ln2_w + l * HDIM, ln2_b + l * HDIM,
                                          nullptr, xbf);
    // MLP1: z1 = gelu(xbf @ W1 + b1) -> bf16   [4096,3072]
    gemm_tn_wmma<<<dim3(H4 / 128, ROWS / 128), blk, 0, stream>>>(
        xbf, W1T + (size_t)l * H4 * HDIM, b1 + l * H4,
        nullptr, z1bf, ROWS, H4, HDIM, /*epi=*/1);
    // MLP2: h += z1 @ W2 + b2  (fused residual) [4096,768]
    gemm_tn_wmma<<<dim3(HDIM / 128, ROWS / 128), blk, 0, stream>>>(
        z1bf, W2T + (size_t)l * HDIM * H4, b2 + l * HDIM,
        h, nullptr, ROWS, HDIM, H4, /*epi=*/2);
  }

  // Final LN -> xbf
  layernorm_k<<<ROWS, blk, 0, stream>>>(h, lnf_w, lnf_b, nullptr, xbf);

  // Head: out = xbf @ embed^T   [4096, 32000]
  gemm_tn_wmma<<<dim3(VOCAB / 128, ROWS / 128), blk, 0, stream>>>(
      xbf, embf, nullptr, out, nullptr, ROWS, VOCAB, HDIM, /*epi=*/0);
}
